// Predictor_87351044866681
// MI455X (gfx1250) — compile-verified, matrix operand order checked by reading, not run on hardware
//
#include <hip/hip_runtime.h>
#include <stdint.h>

// ---------------------------------------------------------------- types
typedef __bf16 bf16_t;
typedef __attribute__((ext_vector_type(8)))  __bf16 v8bf;
typedef __attribute__((ext_vector_type(16))) __bf16 v16bf;
typedef __attribute__((ext_vector_type(8)))  float  v8f;
typedef __attribute__((ext_vector_type(4)))  int    v4i;

#define AS1 __attribute__((address_space(1)))
#define AS3 __attribute__((address_space(3)))

// gfx1250 async global->LDS path (hedged: falls back to VGPR staging if the
// builtin is not declared by this toolchain). Probe round 2 confirmed the
// builtin exists with 4 args and param0 = AS1 int4* (HIP prints it as
// "int __vector_size__(4*sizeof(int)) __device__ *").
#if defined(__gfx1250__) && __has_builtin(__builtin_amdgcn_global_load_async_to_lds_b128)
#define USE_ASYNC_LDS 1
#else
#define USE_ASYNC_LDS 0
#endif

__device__ __forceinline__ void wait_asynccnt0() {
#if defined(__gfx1250__)
#if __has_builtin(__builtin_amdgcn_s_wait_asynccnt)
  __builtin_amdgcn_s_wait_asynccnt(0);
#else
  asm volatile("s_wait_asynccnt 0x0" ::: "memory");
#endif
#endif
}

// ---------------------------------------------------------------- dims
#define DB   16      // batches
#define DM   80      // boxes per batch
#define DRPI 256
#define DN   1280    // DB*DM
#define DR   4096    // DB*DRPI
#define DDF  12544
#define DREP 1024
#define DEMB 256
#define DHID 512
#define DNH  8
#define DL   4
#define DRELC 51

// ---------------------------------------------------------------- helpers
__device__ __forceinline__ float wave_sum(float v) {
#pragma unroll
  for (int m = 16; m > 0; m >>= 1) v += __shfl_xor(v, m, 32);
  return v;
}
__device__ __forceinline__ float wave_max(float v) {
#pragma unroll
  for (int m = 16; m > 0; m >>= 1) v = fmaxf(v, __shfl_xor(v, m, 32));
  return v;
}

// Build a 16x(K=32) bf16 WMMA fragment from an LDS row: elements 0..7 at p,
// elements 8..15 at p+16 (per CDNA5 16-bit A/B VGPR layout).
__device__ __forceinline__ v16bf frag_from_lds(const bf16_t* p) {
  v8bf lo = *(const v8bf*)p;
  v8bf hi = *(const v8bf*)(p + 16);
  v16bf r;
#pragma unroll
  for (int i = 0; i < 8; ++i) { r[i] = lo[i]; r[i + 8] = hi[i]; }
  return r;
}

// ---------------------------------------------------------------- f32 -> bf16
__global__ void k_cvt_bf16(const float* __restrict__ src, bf16_t* __restrict__ dst, int n) {
  int i = blockIdx.x * blockDim.x + threadIdx.x;
  int stride = gridDim.x * blockDim.x;
  for (; i < n; i += stride) dst[i] = (bf16_t)src[i];
}

// ---------------------------------------------------------------- WMMA GEMM
// C[Mr,Nc] = act( A_f32[Mr,K](lda) * W_bf16[Nc,K](ldw)^T + bias ), flags: 1=relu, 2=accumulate-into-C
// grid = (Mr/64, ceil(Nc/64)), block = 256 (8 waves). Mr must be a multiple of 64, K of 32.
__launch_bounds__(256, 2)
__global__ void k_gemm(const float* __restrict__ A, int lda,
                       const bf16_t* __restrict__ W, int ldw,
                       const float* __restrict__ bias,
                       float* __restrict__ C, int ldc,
                       int Nc, int K, int flags) {
  __shared__ bf16_t As[64 * 40];   // 40-elem stride keeps 16B alignment of frag reads
  __shared__ bf16_t Ws[64 * 40];
  const int tid = threadIdx.x;
  const int wid = tid >> 5;
  const int lane = tid & 31;
  const int l15 = lane & 15;
  const int lh  = lane >> 4;
  const int bm = blockIdx.x * 64;
  const int bn = blockIdx.y * 64;
  const int wm = (wid & 3) * 16;   // 4 m-tiles
  const int wn = (wid >> 2) * 32;  // 2 n-groups of 2 tiles

  const int lrow = tid >> 2;       // 64 rows, 4 threads/row
  const int lk   = (tid & 3) * 8;  // 8 elems each

  const float* ap = A + (size_t)(bm + lrow) * lda + lk;
  int wr = bn + lrow; if (wr >= Nc) wr = Nc - 1;
  const bf16_t* wp = W + (size_t)wr * ldw + lk;
  bf16_t* wdst = Ws + lrow * 40 + lk;

  v8f acc0 = {}, acc1 = {};
  const int cn0 = bn + wn + l15;
  const int cn1 = cn0 + 16;
  const int cm  = bm + wm + lh * 8;
  if (flags & 2) {
#pragma unroll
    for (int r = 0; r < 8; ++r) {
      if (cn0 < Nc) acc0[r] = C[(size_t)(cm + r) * ldc + cn0];
      if (cn1 < Nc) acc1[r] = C[(size_t)(cm + r) * ldc + cn1];
    }
  }

  for (int k0 = 0; k0 < K; k0 += 32) {
    float4 f0 = *(const float4*)(ap);
    float4 f1 = *(const float4*)(ap + 4);
    v8bf a8;
    a8[0]=(bf16_t)f0.x; a8[1]=(bf16_t)f0.y; a8[2]=(bf16_t)f0.z; a8[3]=(bf16_t)f0.w;
    a8[4]=(bf16_t)f1.x; a8[5]=(bf16_t)f1.y; a8[6]=(bf16_t)f1.z; a8[7]=(bf16_t)f1.w;
#if !USE_ASYNC_LDS
    v8bf w8 = *(const v8bf*)(wp);
#endif
    if (k0 + 32 < K) {
      __builtin_prefetch(ap + 32, 0, 1);   // global_prefetch_b8
      __builtin_prefetch(wp + 32, 0, 1);
    }
    __syncthreads();                       // all waves done reading previous tiles
#if USE_ASYNC_LDS
    // DMA the already-bf16 weight tile straight into LDS (ASYNCcnt path),
    // while the A tile goes through VALU for the f32->bf16 conversion.
    __builtin_amdgcn_global_load_async_to_lds_b128(
        (AS1 v4i*)(uintptr_t)(const void*)wp,
        (AS3 v4i*)(uint32_t)(uintptr_t)(void*)wdst,
        0, 0);
#endif
    *(v8bf*)(As + lrow * 40 + lk) = a8;
#if USE_ASYNC_LDS
    wait_asynccnt0();                      // weight tile landed in LDS
#else
    *(v8bf*)wdst = w8;
#endif
    __syncthreads();
    v16bf af = frag_from_lds(As + (wm + l15) * 40 + lh * 8);
    v16bf b0 = frag_from_lds(Ws + (wn + l15) * 40 + lh * 8);
    v16bf b1 = frag_from_lds(Ws + (wn + 16 + l15) * 40 + lh * 8);
    acc0 = __builtin_amdgcn_wmma_f32_16x16x32_bf16(false, af, false, b0, (short)0, acc0, false, false);
    acc1 = __builtin_amdgcn_wmma_f32_16x16x32_bf16(false, af, false, b1, (short)0, acc1, false, false);
    ap += 32; wp += 32;
  }

  float bv0 = 0.f, bv1 = 0.f;
  if (bias) { bv0 = bias[cn0 < Nc ? cn0 : 0]; bv1 = bias[cn1 < Nc ? cn1 : 0]; }
#pragma unroll
  for (int r = 0; r < 8; ++r) {
    float v0 = acc0[r] + bv0;
    float v1 = acc1[r] + bv1;
    if (flags & 1) { v0 = fmaxf(v0, 0.f); v1 = fmaxf(v1, 0.f); }
    if (cn0 < Nc) C[(size_t)(cm + r) * ldc + cn0] = v0;
    if (cn1 < Nc) C[(size_t)(cm + r) * ldc + cn1] = v1;
  }
}

// ---------------------------------------------------------------- attention (one block per (b,h))
// qkv: [DN, 3*DHID] f32 (q|k|v, head h at cols h*64). out: [DN, DHID] f32.
__launch_bounds__(256, 2)
__global__ void k_attn(const float* __restrict__ qkv, float* __restrict__ out) {
  __shared__ bf16_t Ks[80 * 64];   // [key][d]
  __shared__ bf16_t Vs[64 * 96];   // [d][key], keys padded to 96
  __shared__ float  Ss[80 * 80];   // scores
  __shared__ bf16_t Ats[80 * 96];  // softmax(att), K padded to 96
  const int b = blockIdx.x >> 3;
  const int h = blockIdx.x & 7;
  const int tid = threadIdx.x, wid = tid >> 5, lane = tid & 31;
  const int l15 = lane & 15, lh = lane >> 4;
  const float* base = qkv + (size_t)(b * 80) * 1536 + h * 64;

  for (int i = tid; i < 80 * 64; i += 256) {
    int n = i >> 6, d = i & 63;
    Ks[n * 64 + d] = (bf16_t)base[n * 1536 + 512 + d];
    Vs[d * 96 + n] = (bf16_t)base[n * 1536 + 1024 + d];
  }
  for (int i = tid; i < 64 * 16; i += 256) {
    int d = i >> 4;
    Vs[d * 96 + 80 + (i & 15)] = (bf16_t)0.f;
  }
  __syncthreads();

  // S = (q k^T) * 1/sqrt(64) : 25 tiles of 16x16, K=64
  for (int t = wid; t < 25; t += 8) {
    int mt = t / 5, nt = t % 5;
    v8f acc = {};
#pragma unroll
    for (int kk = 0; kk < 2; ++kk) {
      const float* qp = base + (size_t)(mt * 16 + l15) * 1536 + kk * 32 + lh * 8;
      float4 q0 = *(const float4*)qp;
      float4 q1 = *(const float4*)(qp + 4);
      float4 q2 = *(const float4*)(qp + 16);
      float4 q3 = *(const float4*)(qp + 20);
      v16bf af;
      af[0]=(bf16_t)q0.x;  af[1]=(bf16_t)q0.y;  af[2]=(bf16_t)q0.z;  af[3]=(bf16_t)q0.w;
      af[4]=(bf16_t)q1.x;  af[5]=(bf16_t)q1.y;  af[6]=(bf16_t)q1.z;  af[7]=(bf16_t)q1.w;
      af[8]=(bf16_t)q2.x;  af[9]=(bf16_t)q2.y;  af[10]=(bf16_t)q2.z; af[11]=(bf16_t)q2.w;
      af[12]=(bf16_t)q3.x; af[13]=(bf16_t)q3.y; af[14]=(bf16_t)q3.z; af[15]=(bf16_t)q3.w;
      v16bf bfrag = frag_from_lds(Ks + (nt * 16 + l15) * 64 + kk * 32 + lh * 8);
      acc = __builtin_amdgcn_wmma_f32_16x16x32_bf16(false, af, false, bfrag, (short)0, acc, false, false);
    }
#pragma unroll
    for (int r = 0; r < 8; ++r)
      Ss[(mt * 16 + lh * 8 + r) * 80 + nt * 16 + l15] = acc[r] * 0.125f;
  }
  __syncthreads();

  // row softmax (one wave per row)
  for (int row = wid; row < 80; row += 8) {
    float mx = -1e30f;
    for (int c = lane; c < 80; c += 32) mx = fmaxf(mx, Ss[row * 80 + c]);
    mx = wave_max(mx);
    float sum = 0.f;
    for (int c = lane; c < 80; c += 32) {
      float e = __expf(Ss[row * 80 + c] - mx);
      Ss[row * 80 + c] = e;
      sum += e;
    }
    sum = wave_sum(sum);
    float inv = 1.f / sum;
    for (int c = lane; c < 96; c += 32)
      Ats[row * 96 + c] = (c < 80) ? (bf16_t)(Ss[row * 80 + c] * inv) : (bf16_t)0.f;
  }
  __syncthreads();

  // O = att @ V : 20 tiles, K=96
  for (int t = wid; t < 20; t += 8) {
    int mt = t / 4, nt = t % 4;
    v8f acc = {};
#pragma unroll
    for (int kk = 0; kk < 3; ++kk) {
      v16bf af    = frag_from_lds(Ats + (mt * 16 + l15) * 96 + kk * 32 + lh * 8);
      v16bf bfrag = frag_from_lds(Vs  + (nt * 16 + l15) * 96 + kk * 32 + lh * 8);
      acc = __builtin_amdgcn_wmma_f32_16x16x32_bf16(false, af, false, bfrag, (short)0, acc, false, false);
    }
#pragma unroll
    for (int r = 0; r < 8; ++r)
      out[(size_t)(b * 80 + mt * 16 + lh * 8 + r) * 512 + h * 64 + nt * 16 + l15] = acc[r];
  }
}

// ---------------------------------------------------------------- add + LayerNorm (D=512), one wave per row
__global__ void k_add_ln(const float* __restrict__ x, const float* __restrict__ y,
                         const float* __restrict__ g, const float* __restrict__ be,
                         float* __restrict__ out, int rows) {
  int row = blockIdx.x * 8 + (threadIdx.x >> 5);
  int lane = threadIdx.x & 31;
  if (row >= rows) return;
  const float* xr = x + (size_t)row * 512;
  const float* yr = y + (size_t)row * 512;
  float v[16];
  float s = 0.f;
#pragma unroll
  for (int i = 0; i < 16; ++i) { v[i] = xr[lane + i * 32] + yr[lane + i * 32]; s += v[i]; }
  s = wave_sum(s) * (1.f / 512.f);
  float var = 0.f;
#pragma unroll
  for (int i = 0; i < 16; ++i) { float d = v[i] - s; var += d * d; }
  var = wave_sum(var) * (1.f / 512.f);
  float inv = rsqrtf(var + 1e-5f);
  float* o = out + (size_t)row * 512;
#pragma unroll
  for (int i = 0; i < 16; ++i) {
    int c = lane + i * 32;
    o[c] = (v[i] - s) * inv * g[c] + be[c];
  }
}

// ---------------------------------------------------------------- box-info MLP (9 -> 32 -> 128), one block per row
__global__ void k_box(const float* __restrict__ boxes,
                      const float* __restrict__ w1, const float* __restrict__ b1,
                      const float* __restrict__ w2, const float* __restrict__ b2,
                      float* __restrict__ pos) {
  __shared__ float bi[9];
  __shared__ float hid[32];
  int row = blockIdx.x, tid = threadIdx.x;
  if (tid == 0) {
    float x1 = boxes[row * 4 + 0], y1 = boxes[row * 4 + 1];
    float x2 = boxes[row * 4 + 2], y2 = boxes[row * 4 + 3];
    float w = x2 - x1 + 1.f, h = y2 - y1 + 1.f;
    float cx = x1 + 0.5f * w, cy = y1 + 0.5f * h;
    bi[0] = w / 800.f;  bi[1] = h / 600.f;
    bi[2] = cx / 800.f; bi[3] = cy / 600.f;
    bi[4] = x1 / 800.f; bi[5] = y1 / 600.f;
    bi[6] = x2 / 800.f; bi[7] = y2 / 600.f;
    bi[8] = w * h / (800.f * 600.f);
  }
  __syncthreads();
  if (tid < 32) {
    float a = b1[tid];
#pragma unroll
    for (int j = 0; j < 9; ++j) a += w1[tid * 9 + j] * bi[j];
    hid[tid] = fmaxf(a, 0.f);
  }
  __syncthreads();
  float a = b2[tid];
#pragma unroll
  for (int j = 0; j < 32; ++j) a += w2[tid * 32 + j] * hid[j];
  pos[(size_t)row * 128 + tid] = fmaxf(a, 0.f);
}

// ---------------------------------------------------------------- gathers / broadcast / mean
__global__ void k_gather_emb(const float* __restrict__ emb, const int* __restrict__ labels,
                             float* __restrict__ out, int n) {  // n = DN*DEMB
  int i = blockIdx.x * blockDim.x + threadIdx.x;
  int stride = gridDim.x * blockDim.x;
  for (; i < n; i += stride) {
    int row = i >> 8, c = i & 255;
    out[i] = emb[(size_t)labels[row] * DEMB + c];
  }
}

__global__ void k_pair(const float* __restrict__ obj_emb, const int* __restrict__ rel,
                       float* __restrict__ out) {  // out [DR, 2*DHID]
  int i = blockIdx.x * blockDim.x + threadIdx.x;
  int stride = gridDim.x * blockDim.x;
  const int n = DR * 2 * DHID;
  for (; i < n; i += stride) {
    int r = i >> 10, c = i & 1023;
    int b = r >> 8, p = r & 255;
    int which = c >> 9;
    int idx = rel[((b << 8) + p) * 2 + which];
    out[i] = obj_emb[(size_t)(b * DM + idx) * DHID + (c & 511)];
  }
}

__global__ void k_mean(const float* __restrict__ x, float* __restrict__ gctx) {  // per-batch mean over DM
  int b = blockIdx.x;
  for (int d = threadIdx.x; d < DHID; d += blockDim.x) {
    float s = 0.f;
    for (int m = 0; m < DM; ++m) s += x[(size_t)(b * DM + m) * DHID + d];
    gctx[(size_t)b * DHID + d] = s * (1.f / (float)DM);
  }
}

__global__ void k_bcast_gctx(const float* __restrict__ gctx, float* __restrict__ out) {  // [DR, DHID]
  int i = blockIdx.x * blockDim.x + threadIdx.x;
  int stride = gridDim.x * blockDim.x;
  const int n = DR * DHID;
  for (; i < n; i += stride) {
    int r = i >> 9;
    out[i] = gctx[(size_t)(r >> 8) * DHID + (i & 511)];
  }
}

// ---------------------------------------------------------------- host
static inline void gemm(hipStream_t s, const float* A, int lda, const bf16_t* W, int ldw,
                        const float* bias, float* C, int ldc, int Mr, int Nc, int K, int flags) {
  dim3 g(Mr / 64, (Nc + 63) / 64);
  k_gemm<<<g, 256, 0, s>>>(A, lda, W, ldw, bias, C, ldc, Nc, K, flags);
}

extern "C" void kernel_launch(void* const* d_in, const int* in_sizes, int n_in,
                              void* d_out, int out_size, void* d_ws, size_t ws_size,
                              hipStream_t stream) {
  (void)in_sizes; (void)n_in; (void)out_size; (void)ws_size;
  const float* features = (const float*)d_in[0];
  const float* unionf   = (const float*)d_in[1];
  const float* boxes    = (const float*)d_in[2];
  const int*   labels   = (const int*)d_in[4];
  const int*   rel      = (const int*)d_in[5];
  const float* fe1_w = (const float*)d_in[6];
  const float* fe1_b = (const float*)d_in[7];
  const float* fe2_w = (const float*)d_in[8];
  const float* fe2_b = (const float*)d_in[9];
  const float* emb1  = (const float*)d_in[10];
  const float* emb2  = (const float*)d_in[11];
  const float* bb1_w = (const float*)d_in[12];
  const float* bb1_b = (const float*)d_in[13];
  const float* bb2_w = (const float*)d_in[14];
  const float* bb2_b = (const float*)d_in[15];
  const float* lobj_w = (const float*)d_in[16];
  const float* lobj_b = (const float*)d_in[17];
  const float* ledge_w = (const float*)d_in[18];
  const float* ledge_b = (const float*)d_in[19];
  const float* pcat_w = (const float*)d_in[44];
  const float* pcat_b = (const float*)d_in[45];
  const float* fuse_w = (const float*)d_in[46];
  const float* fuse_b = (const float*)d_in[47];
  const float* cpx_w  = (const float*)d_in[48];
  const float* cpx_b  = (const float*)d_in[49];
  const float* dpath_w = (const float*)d_in[50];
  const float* dpath_b = (const float*)d_in[51];

  char* ws = (char*)d_ws;
  size_t off = 0;
  auto alloc = [&](size_t bytes) -> void* {
    void* p = ws + off;
    off = (off + bytes + 255) & ~(size_t)255;
    return p;
  };
  auto cvt = [&](const float* src, size_t n) -> bf16_t* {
    bf16_t* dst = (bf16_t*)alloc(n * sizeof(bf16_t));
    int blocks = (int)((n + 255) / 256); if (blocks > 4096) blocks = 4096;
    k_cvt_bf16<<<blocks, 256, 0, stream>>>(src, dst, (int)n);
    return dst;
  };

  // ---- weights -> bf16
  bf16_t* wfe1  = cvt(fe1_w,  (size_t)DREP * DDF);
  bf16_t* wfe2  = cvt(fe2_w,  (size_t)DREP * DREP);
  bf16_t* wlobj = cvt(lobj_w, (size_t)DHID * (DREP + DEMB + 128));
  bf16_t* wledge= cvt(ledge_w,(size_t)DHID * (DEMB + DHID));
  bf16_t* wenc[2][4];
  for (int e = 0; e < 2; ++e) {
    int base = 20 + e * 12;
    wenc[e][0] = cvt((const float*)d_in[base + 0], (size_t)DL * 3 * DHID * DHID); // qkv
    wenc[e][1] = cvt((const float*)d_in[base + 2], (size_t)DL * DHID * DHID);     // out
    wenc[e][2] = cvt((const float*)d_in[base + 4], (size_t)DL * DHID * DHID);     // f1
    wenc[e][3] = cvt((const float*)d_in[base + 6], (size_t)DL * DHID * DHID);     // f2
  }
  bf16_t* wpcat  = cvt(pcat_w,  (size_t)DREP * 2 * DHID);
  bf16_t* wfuse  = cvt(fuse_w,  (size_t)DREP * (DREP + DHID));
  bf16_t* wcpx   = cvt(cpx_w,   (size_t)DRELC * 2 * DREP);
  bf16_t* wdpath = cvt(dpath_w, (size_t)DRELC * DREP);

  // ---- activation buffers
  float* bf1   = (float*)alloc((size_t)DN * DREP * 4);
  float* bf2   = (float*)alloc((size_t)DN * DREP * 4);
  float* pos   = (float*)alloc((size_t)DN * 128 * 4);
  float* e1g   = (float*)alloc((size_t)DN * DEMB * 4);
  float* e2g   = (float*)alloc((size_t)DN * DEMB * 4);
  float* xA    = (float*)alloc((size_t)DN * DHID * 4);
  float* xB    = (float*)alloc((size_t)DN * DHID * 4);
  float* xC    = (float*)alloc((size_t)DN * DHID * 4);
  float* xD    = (float*)alloc((size_t)DN * DHID * 4);
  float* qkvb  = (float*)alloc((size_t)DN * 3 * DHID * 4);
  float* attno = (float*)alloc((size_t)DN * DHID * 4);
  float* ffA   = (float*)alloc((size_t)DN * DHID * 4);
  float* ffB   = (float*)alloc((size_t)DN * DHID * 4);
  float* gctx  = (float*)alloc((size_t)DB * DHID * 4);
  float* gexp  = (float*)alloc((size_t)DR * DHID * 4);
  float* pair  = (float*)alloc((size_t)DR * 2 * DHID * 4);
  float* local = (float*)alloc((size_t)DR * DREP * 4);
  float* sem   = (float*)alloc((size_t)DR * DREP * 4);
  float* outC  = (float*)d_out;

  // ---- feature MLP
  gemm(stream, features, DDF, wfe1, DDF, fe1_b, bf1, DREP, DN, DREP, DDF, 1);
  gemm(stream, bf1, DREP, wfe2, DREP, fe2_b, bf2, DREP, DN, DREP, DREP, 1);

  // ---- box MLP + embedding gathers
  k_box<<<DN, 128, 0, stream>>>(boxes, bb1_w, bb1_b, bb2_w, bb2_b, pos);
  {
    int n = DN * DEMB, blocks = (n + 255) / 256;
    k_gather_emb<<<blocks, 256, 0, stream>>>(emb1, labels, e1g, n);
    k_gather_emb<<<blocks, 256, 0, stream>>>(emb2, labels, e2g, n);
  }

  // ---- obj_rep = [bf2 | emb1[labels] | pos] @ lobj_w^T + b  (concat via accumulate passes)
  gemm(stream, bf2, DREP, wlobj, DREP + DEMB + 128, lobj_b, xA, DHID, DN, DHID, DREP, 0);
  gemm(stream, e1g, DEMB, wlobj + DREP, DREP + DEMB + 128, nullptr, xA, DHID, DN, DHID, DEMB, 2);
  gemm(stream, pos, 128,  wlobj + DREP + DEMB, DREP + DEMB + 128, nullptr, xA, DHID, DN, DHID, 128, 2);

  // ---- transformer encoder (shared for both)
  auto enc = [&](int e, float* cur, float* alt) {
    int base = 20 + e * 12;
    const float* bqkv = (const float*)d_in[base + 1];
    const float* bout = (const float*)d_in[base + 3];
    const float* bF1  = (const float*)d_in[base + 5];
    const float* bF2  = (const float*)d_in[base + 7];
    const float* l1s  = (const float*)d_in[base + 8];
    const float* l1b  = (const float*)d_in[base + 9];
    const float* l2s  = (const float*)d_in[base + 10];
    const float* l2b  = (const float*)d_in[base + 11];
    for (int l = 0; l < DL; ++l) {
      gemm(stream, cur, DHID, wenc[e][0] + (size_t)l * 3 * DHID * DHID, DHID,
           bqkv + l * 3 * DHID, qkvb, 3 * DHID, DN, 3 * DHID, DHID, 0);
      k_attn<<<DB * DNH, 256, 0, stream>>>(qkvb, attno);
      gemm(stream, attno, DHID, wenc[e][1] + (size_t)l * DHID * DHID, DHID,
           bout + l * DHID, ffA, DHID, DN, DHID, DHID, 0);
      k_add_ln<<<DN / 8, 256, 0, stream>>>(cur, ffA, l1s + l * DHID, l1b + l * DHID, alt, DN);
      gemm(stream, alt, DHID, wenc[e][2] + (size_t)l * DHID * DHID, DHID,
           bF1 + l * DHID, ffA, DHID, DN, DHID, DHID, 1);
      gemm(stream, ffA, DHID, wenc[e][3] + (size_t)l * DHID * DHID, DHID,
           bF2 + l * DHID, ffB, DHID, DN, DHID, DHID, 0);
      k_add_ln<<<DN / 8, 256, 0, stream>>>(alt, ffB, l2s + l * DHID, l2b + l * DHID, cur, DN);
    }
  };

  enc(0, xA, xB);                    // obj encoder; obj_emb = xA

  // ---- edge_in = [obj_emb | emb2[labels]] @ ledge_w^T + b
  gemm(stream, xA,  DHID, wledge, DEMB + DHID, ledge_b, xC, DHID, DN, DHID, DHID, 0);
  gemm(stream, e2g, DEMB, wledge + DHID, DEMB + DHID, nullptr, xC, DHID, DN, DHID, DEMB, 2);

  enc(1, xC, xD);                    // edge encoder; gctx_full = xC
  k_mean<<<DB, 256, 0, stream>>>(xC, gctx);
  {
    int n = DR * DHID, blocks = (n + 255) / 256;
    k_bcast_gctx<<<blocks, 256, 0, stream>>>(gctx, gexp);
  }

  // ---- pair gather + local/sem
  {
    int n = DR * 2 * DHID, blocks = (n + 255) / 256; if (blocks > 8192) blocks = 8192;
    k_pair<<<blocks, 256, 0, stream>>>(xA, rel, pair);
  }
  gemm(stream, pair, 2 * DHID, wpcat, 2 * DHID, pcat_b, local, DREP, DR, DREP, 2 * DHID, 0);
  // sem = relu(local @ fuse_w[:, :1024]^T + gexp @ fuse_w[:, 1024:]^T + b)
  gemm(stream, local, DREP, wfuse, DREP + DHID, fuse_b, sem, DREP, DR, DREP, DREP, 0);
  gemm(stream, gexp, DHID, wfuse + DREP, DREP + DHID, nullptr, sem, DREP, DR, DREP, DHID, 2 | 1);

  // ---- rel_logits = [sem | union] @ cpx_w^T + cpx_b + local @ dpath_w^T + dpath_b
  gemm(stream, sem, DREP, wcpx, 2 * DREP, cpx_b, outC, DRELC, DR, DRELC, DREP, 0);
  gemm(stream, unionf, DREP, wcpx + DREP, 2 * DREP, nullptr, outC, DRELC, DR, DRELC, DREP, 2);
  gemm(stream, local, DREP, wdpath, DREP, dpath_b, outC, DRELC, DR, DRELC, DREP, 2);
}
